// HAN_61280593379920
// MI455X (gfx1250) — matrix operand order repaired and checked by reading, not compiled, not run on hardware
//
#include <hip/hip_runtime.h>
#include <hip/hip_bf16.h>
#include <math.h>

// MI455X / gfx1250, wave32. WMMA f16->f32 for the three feature GEMMs,
// L2-resident [N,64] intermediates, global f32 atomics for GCN scatter.

typedef __attribute__((ext_vector_type(16))) _Float16 v16h;
typedef __attribute__((ext_vector_type(8)))  _Float16 v8h;
typedef __attribute__((ext_vector_type(8)))  float    v8f;
typedef __attribute__((ext_vector_type(4)))  float    v4f;

#define HID 64

__device__ __forceinline__ float elu1(float v) {
    return v > 0.0f ? v : (__expf(v) - 1.0f);
}

__device__ __forceinline__ void atomAddF(float* p, float v) {
    __hip_atomic_fetch_add(p, v, __ATOMIC_RELAXED, __HIP_MEMORY_SCOPE_AGENT);
}

// ---------------------------------------------------------------------------
// WMMA GEMM:  out[M,64] = A[M,KDIM] @ W[64,KDIM]^T   (+ optional fused work)
// FUSE: A element = 0.5 + elu(A + bin[k]) + elu(coef * A2)   (xres1 on the fly)
// EPI : out = elu(acc + bepi[col])
// Block = 256 threads = 8 waves; wave w owns rows [blk*128 + w*16, +16).
// ---------------------------------------------------------------------------
template<int KDIM, bool FUSE, bool EPI>
__global__ __launch_bounds__(256) void gemm_k(
    const float* __restrict__ A,     const float* __restrict__ A2,
    const float* __restrict__ bin,   const float* __restrict__ coefp,
    const float* __restrict__ Wm,    const float* __restrict__ bepi,
    float* __restrict__ out, int M)
{
    constexpr int LST = KDIM + 8;                 // pad: 16B-aligned rows, no bank conflicts
    __shared__ __align__(16) _Float16 Wh[HID * LST];
    __shared__ float binl[FUSE ? KDIM : 1];

    const int tid = threadIdx.x;
    for (int i = tid; i < HID * KDIM; i += 256) {
        int r = i / KDIM, c = i % KDIM;           // KDIM is a power of two
        Wh[r * LST + c] = (_Float16)Wm[i];
    }
    if constexpr (FUSE) {
        for (int i = tid; i < KDIM; i += 256) binl[i] = bin[i];
    }
    __syncthreads();

    const float coef = FUSE ? coefp[0] : 0.0f;
    const int lane = tid & 31;
    const int wv   = tid >> 5;
    const int row0 = blockIdx.x * 128 + wv * 16;
    const int half = lane >> 4;
    const int mrow = row0 + (lane & 15);
    const int mcl  = mrow < M ? mrow : (M - 1);   // clamp loads, guard stores (EXEC stays all-1)
    const float* ap  = A + (size_t)mcl * KDIM;
    const float* a2p = FUSE ? (A2 + (size_t)mcl * KDIM) : A;

    v8f acc[4] = {};

    #pragma unroll
    for (int kc = 0; kc < KDIM / 32; ++kc) {
        // 16-bit A layout: lane half=0 -> K = [kb..kb+7] U [kb+16..kb+23], half=1 -> +8
        const int kb1 = kc * 32 + half * 8;
        const int kb2 = kb1 + 16;
        v4f q0 = *(const v4f*)(ap + kb1);
        v4f q1 = *(const v4f*)(ap + kb1 + 4);
        v4f q2 = *(const v4f*)(ap + kb2);
        v4f q3 = *(const v4f*)(ap + kb2 + 4);
        float f[16];
        #pragma unroll
        for (int j = 0; j < 4; ++j) { f[j]=q0[j]; f[4+j]=q1[j]; f[8+j]=q2[j]; f[12+j]=q3[j]; }
        if constexpr (FUSE) {
            v4f p0 = *(const v4f*)(a2p + kb1);
            v4f p1 = *(const v4f*)(a2p + kb1 + 4);
            v4f p2 = *(const v4f*)(a2p + kb2);
            v4f p3 = *(const v4f*)(a2p + kb2 + 4);
            float g[16];
            #pragma unroll
            for (int j = 0; j < 4; ++j) { g[j]=p0[j]; g[4+j]=p1[j]; g[8+j]=p2[j]; g[12+j]=p3[j]; }
            #pragma unroll
            for (int j = 0; j < 16; ++j) {
                int k = (j < 8) ? (kb1 + j) : (kb2 + j - 8);
                f[j] = 0.5f + elu1(f[j] + binl[k]) + elu1(coef * g[j]);
            }
        }
        v16h a;
        #pragma unroll
        for (int j = 0; j < 16; ++j) a[j] = (_Float16)f[j];

        const int boff = kc * 32 + half * 8;      // B: halves [boff..+7] U [boff+16..+23] of W row n
        #pragma unroll
        for (int nt = 0; nt < 4; ++nt) {
            const _Float16* bp = &Wh[(nt * 16 + (lane & 15)) * LST + boff];
            v8h blo = *(const v8h*)bp;
            v8h bhi = *(const v8h*)(bp + 16);
            v16h b = __builtin_shufflevector(blo, bhi,
                       0,1,2,3,4,5,6,7,8,9,10,11,12,13,14,15);
            acc[nt] = __builtin_amdgcn_wmma_f32_16x16x32_f16(
                false, a, false, b, (short)0, acc[nt], false, false);
        }
    }

    #pragma unroll
    for (int nt = 0; nt < 4; ++nt) {
        const int col = nt * 16 + (lane & 15);
        const float badd = EPI ? bepi[col] : 0.0f;
        #pragma unroll
        for (int r = 0; r < 8; ++r) {
            const int row = row0 + half * 8 + r;  // f32 C/D layout
            if (row < M) {
                float v = acc[nt][r];
                if constexpr (EPI) v = elu1(v + badd);
                out[(size_t)row * HID + col] = v;
            }
        }
    }
}

// ---------------------------------------------------------------------------
// Semantic attention over the weight matrix itself -> single scalar
// coefOut = ALPHA * att[0,0].  One block of 128 threads (ATT_HID).
// ---------------------------------------------------------------------------
__global__ void attn_scalar_k(const float* __restrict__ W, int C,
                              const float* __restrict__ aw1,
                              const float* __restrict__ ab1,
                              const float* __restrict__ aw2,
                              float* __restrict__ coefOut)
{
    __shared__ float red[128];
    __shared__ float s[HID];
    const int h = threadIdx.x;                    // 0..127
    for (int r = 0; r < HID; ++r) {
        float dot = ab1[h];
        for (int c = 0; c < C; ++c) dot += W[r * C + c] * aw1[h * C + c];
        red[h] = aw2[h] * tanhf(dot);
        __syncthreads();
        for (int o = 64; o > 0; o >>= 1) {
            if (h < o) red[h] += red[h + o];
            __syncthreads();
        }
        if (h == 0) s[r] = red[0];
        __syncthreads();
    }
    if (h == 0) {
        float mx = s[0];
        for (int r = 1; r < HID; ++r) mx = fmaxf(mx, s[r]);
        float sum = 0.0f;
        for (int r = 0; r < HID; ++r) { float e = __expf(s[r] - mx); s[r] = e; sum += e; }
        float a00 = 0.0f;
        for (int r = 0; r < HID; ++r) a00 += (s[r] / sum) * W[r * C];
        coefOut[0] = 0.5f * a00;                  // ALPHA = 0.5
    }
}

// --------------------------- small elementwise kernels ----------------------
__global__ void fill1_k(float* __restrict__ p, int n) {
    int i = blockIdx.x * 256 + threadIdx.x;
    if (i < n) p[i] = 1.0f;                       // self-loop contribution to degree
}
__global__ void degcount_k(const int* __restrict__ dst, float* __restrict__ deg, int E) {
    int e = blockIdx.x * 256 + threadIdx.x;
    if (e < E) atomAddF(&deg[dst[e]], 1.0f);
}
__global__ void rsqrt_k(float* __restrict__ p, int n) {
    int i = blockIdx.x * 256 + threadIdx.x;
    if (i < n) p[i] = rsqrtf(p[i]);               // deg >= 1 always
}
__global__ void selfloop_k(const float* __restrict__ xlin, const float* __restrict__ dinv,
                           float* __restrict__ hacc, int total) {
    int i = blockIdx.x * 256 + threadIdx.x;
    if (i < total) { float di = dinv[i >> 6]; hacc[i] = di * di * xlin[i]; }
}
__global__ void edge_scatter_k(const float* __restrict__ xlin, const int* __restrict__ src,
                               const int* __restrict__ dst, const float* __restrict__ dinv,
                               float* __restrict__ hacc, int E) {
    int t = blockIdx.x * 256 + threadIdx.x;       // E*64 = 51.2M, fits int
    int e = t >> 6, f = t & 63;
    if (e < E) {
        int s = src[e], d = dst[e];
        float w = dinv[s] * dinv[d];
        atomAddF(&hacc[(size_t)d * HID + f], xlin[(size_t)s * HID + f] * w);
    }
}
__global__ void xres2_k(const float* __restrict__ hacc, const float* __restrict__ xlin,
                        const float* __restrict__ b2, const float* __restrict__ coefp,
                        float* __restrict__ out, int total) {
    int i = blockIdx.x * 256 + threadIdx.x;
    if (i < total) {
        int f = i & 63;
        out[i] = 0.5f + hacc[i] + b2[f] + elu1(coefp[0] * xlin[i]);
    }
}
// One wave32 per edge pair: lane f + f+32, butterfly reduce.
__global__ void edge_score_k(const float* __restrict__ xr, const int* __restrict__ pos,
                             const int* __restrict__ neg, float* __restrict__ out, int ELP) {
    int gt = blockIdx.x * 256 + threadIdx.x;
    int wid = gt >> 5, lane = gt & 31;
    if (wid >= 2 * ELP) return;
    const int* pp = (wid < ELP) ? (pos + 2 * (size_t)wid) : (neg + 2 * (size_t)(wid - ELP));
    int a = pp[0], b = pp[1];
    const float* xa = xr + (size_t)a * HID;
    const float* xb = xr + (size_t)b * HID;
    float sv = xa[lane] * xb[lane] + xa[lane + 32] * xb[lane + 32];
    #pragma unroll
    for (int o = 16; o > 0; o >>= 1) sv += __shfl_xor(sv, o, 32);
    if (lane == 0) out[wid] = sv;
}

// ---------------------------------------------------------------------------
extern "C" void kernel_launch(void* const* d_in, const int* in_sizes, int n_in,
                              void* d_out, int out_size, void* d_ws, size_t ws_size,
                              hipStream_t stream)
{
    const float* x    = (const float*)d_in[0];
    const int*   ei   = (const int*)d_in[1];
    const int*   epos = (const int*)d_in[2];
    const int*   eneg = (const int*)d_in[3];
    const float* W1   = (const float*)d_in[4];
    const float* b1   = (const float*)d_in[5];
    const float* a1w1 = (const float*)d_in[6];
    const float* a1b1 = (const float*)d_in[7];
    const float* a1w2 = (const float*)d_in[8];
    const float* cw1  = (const float*)d_in[9];
    const float* cb1  = (const float*)d_in[10];
    const float* W2   = (const float*)d_in[11];
    const float* b2   = (const float*)d_in[12];
    const float* a2w1 = (const float*)d_in[13];
    const float* a2b1 = (const float*)d_in[14];
    const float* a2w2 = (const float*)d_in[15];

    const int N   = in_sizes[0] / 128;
    const int E   = in_sizes[1] / 2;
    const int ELP = in_sizes[2] / 2;
    const int* src = ei;
    const int* dst = ei + E;

    float* ws = (float*)d_ws;
    const size_t NH = (size_t)N * HID;
    float* xlin = ws;                 // xlin1, later reused as xlin2
    float* hacc = ws + NH;            // h1 accumulator, later h2
    float* x2   = ws + 2 * NH;        // x2, later xres2
    float* dinv = ws + 3 * NH;        // degree -> d^-1/2
    float* coef = ws + 3 * NH + N;    // [coef1, coef2]

    dim3 B(256);
    fill1_k   <<<(N + 255) / 256, B, 0, stream>>>(dinv, N);
    degcount_k<<<(E + 255) / 256, B, 0, stream>>>(dst, dinv, E);
    rsqrt_k   <<<(N + 255) / 256, B, 0, stream>>>(dinv, N);
    attn_scalar_k<<<1, 128, 0, stream>>>(W1, 128, a1w1, a1b1, a1w2, coef);
    attn_scalar_k<<<1, 128, 0, stream>>>(W2, 64,  a2w1, a2b1, a2w2, coef + 1);

    const int gb  = (N + 127) / 128;
    const int tot = N * HID;
    const int etot = E * HID;

    // view 1: xlin1 = x @ W1^T
    gemm_k<128, false, false><<<gb, B, 0, stream>>>(x, nullptr, nullptr, nullptr,
                                                    W1, nullptr, xlin, N);
    selfloop_k    <<<(tot + 255) / 256, B, 0, stream>>>(xlin, dinv, hacc, tot);
    edge_scatter_k<<<(etot + 255) / 256, B, 0, stream>>>(xlin, src, dst, dinv, hacc, E);
    // x2 = elu(xres1 @ cw1^T + cb1), xres1 fused into A-load
    gemm_k<64, true, true><<<gb, B, 0, stream>>>(hacc, xlin, b1, coef,
                                                 cw1, cb1, x2, N);
    // view 2: xlin2 = x2 @ W2^T (overwrite xlin buffer)
    gemm_k<64, false, false><<<gb, B, 0, stream>>>(x2, nullptr, nullptr, nullptr,
                                                   W2, nullptr, xlin, N);
    selfloop_k    <<<(tot + 255) / 256, B, 0, stream>>>(xlin, dinv, hacc, tot);
    edge_scatter_k<<<(etot + 255) / 256, B, 0, stream>>>(xlin, src, dst, dinv, hacc, E);
    // xres2 into x2 buffer (x2 dead after GEMM3)
    xres2_k<<<(tot + 255) / 256, B, 0, stream>>>(hacc, xlin, b2, coef + 1, x2, tot);
    // edge scoring
    const long long sthreads = 2LL * ELP * 32;
    edge_score_k<<<(unsigned)((sthreads + 255) / 256), B, 0, stream>>>(
        x2, epos, eneg, (float*)d_out, ELP);

    (void)n_in; (void)out_size; (void)ws_size;
}
